// Block_90254442758889
// MI455X (gfx1250) — matrix operand order repaired and checked by reading, not compiled, not run on hardware
//
#include <hip/hip_runtime.h>

// ---------------------------------------------------------------------------
// CDNA5 (gfx1250) wave32 WMMA transformer block
//   b=1, t=4096, H=2048, qh=16, kh=4, d=128, F=8192
// ---------------------------------------------------------------------------

typedef __attribute__((ext_vector_type(16))) __bf16 bf16x16;
typedef __attribute__((ext_vector_type(8)))  float  f32x8;
typedef __attribute__((ext_vector_type(4)))  int    v4i;

union Frag { bf16x16 v; uint4 q[2]; };

// ---- optional CDNA5 async Global->LDS path (guarded, falls back to uint4) --
#if defined(__gfx1250__) && \
    __has_builtin(__builtin_amdgcn_global_load_async_to_lds_b128) && \
    __has_builtin(__builtin_amdgcn_s_wait_asynccnt)
#define CDNA5_ASYNC 1
#else
#define CDNA5_ASYNC 0
#endif

static __device__ inline void cp_b128(unsigned short* dst_lds,
                                      const unsigned short* src_g) {
#if CDNA5_ASYNC
    __builtin_amdgcn_global_load_async_to_lds_b128(
        (__attribute__((address_space(1))) v4i*)src_g,
        (__attribute__((address_space(3))) v4i*)dst_lds, 0, 0);
#else
    *(uint4*)dst_lds = *(const uint4*)src_g;
#endif
}
static __device__ inline void cp_wait() {
#if CDNA5_ASYNC
    __builtin_amdgcn_s_wait_asynccnt(0);
#endif
}

static __device__ inline float bf2f(unsigned short u) {
    unsigned int x = ((unsigned int)u) << 16;
    float f; __builtin_memcpy(&f, &x, 4); return f;
}
static __device__ inline unsigned short f2bf(float f) {
    unsigned int x; __builtin_memcpy(&x, &f, 4);
    x += 0x7FFFu + ((x >> 16) & 1u);          // round-to-nearest-even
    return (unsigned short)(x >> 16);
}
static __device__ inline void unpack8(uint4 p, float* v) {
    unsigned int w[4] = {p.x, p.y, p.z, p.w};
#pragma unroll
    for (int i = 0; i < 4; ++i) {
        v[2*i]   = bf2f((unsigned short)(w[i] & 0xffffu));
        v[2*i+1] = bf2f((unsigned short)(w[i] >> 16));
    }
}
static __device__ inline uint4 pack8(const unsigned short* o) {
    uint4 p;
    p.x = (unsigned int)o[0] | ((unsigned int)o[1] << 16);
    p.y = (unsigned int)o[2] | ((unsigned int)o[3] << 16);
    p.z = (unsigned int)o[4] | ((unsigned int)o[5] << 16);
    p.w = (unsigned int)o[6] | ((unsigned int)o[7] << 16);
    return p;
}
static __device__ inline float wave_sum(float v) {
#pragma unroll
    for (int o = 16; o > 0; o >>= 1) v += __shfl_xor(v, o, 32);
    return v;
}

// ---------------------------------------------------------------------------
// RMSNorm over rows of 2048 (one block per row, 256 threads, 8 elems/thread)
// ---------------------------------------------------------------------------
__global__ __launch_bounds__(256)
void rmsnorm2048_kernel(const unsigned short* __restrict__ x,
                        const float* __restrict__ gamma,
                        unsigned short* __restrict__ out) {
    const int row = blockIdx.x, tid = threadIdx.x;
    const unsigned short* xr = x + (size_t)row * 2048;
    float v[8];
    unpack8(*(const uint4*)(xr + tid * 8), v);
    float ss = 0.f;
#pragma unroll
    for (int r = 0; r < 8; ++r) ss += v[r] * v[r];
    ss = wave_sum(ss);
    __shared__ float red[8];
    if ((tid & 31) == 0) red[tid >> 5] = ss;
    __syncthreads();
    float tot = 0.f;
#pragma unroll
    for (int i = 0; i < 8; ++i) tot += red[i];
    const float rinv = rsqrtf(tot * (1.0f / 2048.0f) + 1e-6f);
    unsigned short o[8];
#pragma unroll
    for (int r = 0; r < 8; ++r) o[r] = f2bf(gamma[tid * 8 + r] * v[r] * rinv);
    *(uint4*)(out + (size_t)row * 2048 + tid * 8) = pack8(o);
}

// ---------------------------------------------------------------------------
// Tiled bf16 WMMA GEMM: C[M,N] = A[M,K] @ B[K,N] (+Res), bf16 in/out, f32 acc
// Block 256 thr = 8 waves; tile 256x128, BK=64; wave = 64x64 patch (4x4 tiles)
// -> 16 v_wmma per 16 ds_load_b128 per 32-wide K chunk.
// ---------------------------------------------------------------------------
#define GLDA 72   // LDS k-stride (elems): 144B, 16B aligned, bank-skewed

template<bool RES>
__global__ __launch_bounds__(256)
void gemm_bf16(const unsigned short* __restrict__ A,
               const unsigned short* __restrict__ B,
               const unsigned short* __restrict__ Res,
               unsigned short* __restrict__ C,
               int M, int N, int K) {
    __shared__ unsigned short As[256 * GLDA];
    __shared__ unsigned short Bt[128 * GLDA];   // transposed: [n][k]
    const int tid = threadIdx.x, lane = tid & 31, w = tid >> 5;
    const int wr = w >> 1, wc = w & 1;          // wave grid 4x2 of 64x64
    const int l15 = lane & 15, lhi = lane >> 4;
    const int akb = lhi * 8, bkb = lhi * 8;
    const int m0 = blockIdx.y * 256, n0 = blockIdx.x * 128;

    f32x8 acc[4][4] = {};

    for (int k0 = 0; k0 < K; k0 += 64) {
        __syncthreads();
#pragma unroll
        for (int it = 0; it < 8; ++it) {            // A: 256x64, k-contiguous
            int idx = tid + it * 256;
            int r = idx >> 3, c8 = (idx & 7) * 8;
            cp_b128(As + r * GLDA + c8, A + (size_t)(m0 + r) * K + k0 + c8);
        }
#pragma unroll
        for (int it = 0; it < 4; ++it) {            // B: 64x128 -> Bt[n][k]
            int idx = tid + it * 256;
            int kk = idx >> 4, g = (idx & 15) * 8;
            uint4 dv = *(const uint4*)(B + (size_t)(k0 + kk) * N + n0 + g);
            unsigned int wv4[4] = {dv.x, dv.y, dv.z, dv.w};
#pragma unroll
            for (int e = 0; e < 4; ++e) {
                Bt[(g + 2*e    ) * GLDA + kk] = (unsigned short)(wv4[e] & 0xffffu);
                Bt[(g + 2*e + 1) * GLDA + kk] = (unsigned short)(wv4[e] >> 16);
            }
        }
        if (k0 + 64 < K)                            // stream next weight tile
            __builtin_prefetch(B + (size_t)(k0 + 64 + (tid >> 4)) * N + n0 + (tid & 15) * 8, 0, 0);
        cp_wait();
        __syncthreads();
#pragma unroll
        for (int kc = 0; kc < 64; kc += 32) {
            Frag a[4], b[4];
#pragma unroll
            for (int mi = 0; mi < 4; ++mi) {
                const unsigned short* p = As + (64*wr + 16*mi + l15) * GLDA + kc + akb;
                a[mi].q[0] = *(const uint4*)p;
                a[mi].q[1] = *(const uint4*)(p + 16);
            }
#pragma unroll
            for (int ni = 0; ni < 4; ++ni) {
                const unsigned short* p = Bt + (64*wc + 16*ni + l15) * GLDA + kc + bkb;
                b[ni].q[0] = *(const uint4*)p;
                b[ni].q[1] = *(const uint4*)(p + 16);
            }
#pragma unroll
            for (int mi = 0; mi < 4; ++mi)
#pragma unroll
                for (int ni = 0; ni < 4; ++ni)
                    acc[mi][ni] = __builtin_amdgcn_wmma_f32_16x16x32_bf16(
                        false, a[mi].v, false, b[ni].v, (short)0, acc[mi][ni],
                        false, false);
        }
    }
#pragma unroll
    for (int mi = 0; mi < 4; ++mi)
#pragma unroll
        for (int ni = 0; ni < 4; ++ni) {
            int n = n0 + 64*wc + 16*ni + l15;
#pragma unroll
            for (int r = 0; r < 8; ++r) {
                int m = m0 + 64*wr + 16*mi + (lhi ? 8 : 0) + r;
                float vv = acc[mi][ni][r];
                if (RES) vv += bf2f(Res[(size_t)m * N + n]);
                C[(size_t)m * N + n] = f2bf(vv);
            }
        }
}

// ---------------------------------------------------------------------------
// Per-head (d=128) RMSNorm + RoPE; optional f32 copy (returned k)
// One block (128 thr) per (t, head) row; in-place safe.
// ---------------------------------------------------------------------------
__global__ __launch_bounds__(128)
void rope_rmsnorm_kernel(const unsigned short* __restrict__ in,
                         const float* __restrict__ gamma,
                         const float* __restrict__ sinp,
                         const float* __restrict__ cosp,
                         unsigned short* __restrict__ out_bf,
                         float* __restrict__ out_f32,
                         int nheads) {
    const int row = blockIdx.x, i = threadIdx.x;
    const int tpos = row / nheads;
    float f = bf2f(in[(size_t)row * 128 + i]);
    float ss = wave_sum(f * f);
    __shared__ float red[4];
    __shared__ float nsh[128];
    if ((i & 31) == 0) red[i >> 5] = ss;
    __syncthreads();
    const float tot = red[0] + red[1] + red[2] + red[3];
    const float rinv = rsqrtf(tot * (1.0f / 128.0f) + 1e-6f);
    nsh[i] = bf2f(f2bf(gamma[i] * f * rinv));   // bf16 round like reference
    __syncthreads();
    const int hh = i & 63;
    const float s = sinp[(size_t)tpos * 64 + hh];
    const float c = cosp[(size_t)tpos * 64 + hh];
    float o = (i < 64) ? (nsh[i] * c - nsh[i + 64] * s)
                       : (nsh[i] * c + nsh[i - 64] * s);
    out_bf[(size_t)row * 128 + i] = f2bf(o);
    if (out_f32) out_f32[(size_t)row * 128 + i] = o;
}

// ---------------------------------------------------------------------------
// Causal GQA flash attention, WMMA bf16. grid = (t/64, qh); 256 thr = 8 waves.
// Q [t][16*128], K/V [t][4*128]. Wave (wr,wc): S rows 16wr cols 32wc,
// O rows 16wr cols 64wc (4 tiles over d=128).
// ---------------------------------------------------------------------------
__global__ __launch_bounds__(256)
void flash_attn_kernel(const unsigned short* __restrict__ Q,
                       const unsigned short* __restrict__ Kb,
                       const unsigned short* __restrict__ Vb,
                       unsigned short* __restrict__ O) {
    __shared__ unsigned short Qs[64 * 136];
    __shared__ unsigned short Ks[64 * 136];
    __shared__ unsigned short Vt[128 * 72];    // transposed [d][kn]
    __shared__ float          Sf[64 * 68];
    __shared__ unsigned short Pb[64 * 72];
    __shared__ float alpha_s[64], linv_s[64];

    const int tid = threadIdx.x, lane = tid & 31, w = tid >> 5;
    const int wr = w >> 1, wc = w & 1, l15 = lane & 15, lhi = lane >> 4;
    const int i0 = blockIdx.x * 64;
    const int h = blockIdx.y, kvh = h >> 2;
    const float scale = 0.08838834764831845f;  // 1/sqrt(128)

#pragma unroll
    for (int it = 0; it < 4; ++it) {           // stage Q tile once (async)
        int idx = tid + it * 256;
        int r = idx >> 4, g = (idx & 15) * 8;
        cp_b128(Qs + r * 136 + g, Q + ((size_t)(i0 + r) * 16 + h) * 128 + g);
    }

    f32x8 o[4] = {};
    float m_i = -1e30f, l_i = 0.0f;            // live in threads 0..63

    const int jmax = blockIdx.x;
    for (int j = 0; j <= jmax; ++j) {
        __syncthreads();
#pragma unroll
        for (int it = 0; it < 4; ++it) {       // K tile, row-major (async)
            int idx = tid + it * 256;
            int r = idx >> 4, g = (idx & 15) * 8;
            cp_b128(Ks + r * 136 + g,
                    Kb + ((size_t)(j * 64 + r) * 4 + kvh) * 128 + g);
        }
#pragma unroll
        for (int it = 0; it < 4; ++it) {       // V tile, transposed to Vt
            int idx = tid + it * 256;
            int r = idx >> 4, g = (idx & 15) * 8;
            uint4 dv = *(const uint4*)(Vb + ((size_t)(j * 64 + r) * 4 + kvh) * 128 + g);
            unsigned int wv4[4] = {dv.x, dv.y, dv.z, dv.w};
#pragma unroll
            for (int e = 0; e < 4; ++e) {
                Vt[(g + 2*e    ) * 72 + r] = (unsigned short)(wv4[e] & 0xffffu);
                Vt[(g + 2*e + 1) * 72 + r] = (unsigned short)(wv4[e] >> 16);
            }
        }
        cp_wait();
        __syncthreads();

        f32x8 s[2] = {};
#pragma unroll
        for (int kc = 0; kc < 128; kc += 32) { // S = Q K^T
            Frag a, b0, b1;
            const unsigned short* pa = Qs + (16*wr + l15) * 136 + kc + lhi * 8;
            a.q[0] = *(const uint4*)pa;  a.q[1] = *(const uint4*)(pa + 16);
            const unsigned short* pb0 = Ks + (32*wc + l15) * 136 + kc + lhi * 8;
            b0.q[0] = *(const uint4*)pb0; b0.q[1] = *(const uint4*)(pb0 + 16);
            const unsigned short* pb1 = Ks + (32*wc + 16 + l15) * 136 + kc + lhi * 8;
            b1.q[0] = *(const uint4*)pb1; b1.q[1] = *(const uint4*)(pb1 + 16);
            s[0] = __builtin_amdgcn_wmma_f32_16x16x32_bf16(false, a.v, false, b0.v, (short)0, s[0], false, false);
            s[1] = __builtin_amdgcn_wmma_f32_16x16x32_bf16(false, a.v, false, b1.v, (short)0, s[1], false, false);
        }
#pragma unroll
        for (int ti = 0; ti < 2; ++ti) {       // scale + causal mask -> Sf
            int n = 32*wc + 16*ti + l15;
            int kcol = j * 64 + n;
#pragma unroll
            for (int r = 0; r < 8; ++r) {
                int m = 16*wr + (lhi ? 8 : 0) + r;
                float vle = s[ti][r] * scale;
                if (kcol > i0 + m) vle = -1e30f;
                Sf[m * 68 + n] = vle;
            }
        }
        __syncthreads();
        if (tid < 64) {                        // online softmax, row = tid
            float mx = m_i;
            for (int n = 0; n < 64; ++n) mx = fmaxf(mx, Sf[tid * 68 + n]);
            float al = __expf(m_i - mx);
            float ls = 0.f;
            for (int n = 0; n < 64; ++n) {
                float p = __expf(Sf[tid * 68 + n] - mx);
                Pb[tid * 72 + n] = f2bf(p);
                ls += p;
            }
            l_i = l_i * al + ls;
            m_i = mx;
            alpha_s[tid] = al;
        }
        __syncthreads();
        float av[8];
#pragma unroll
        for (int r = 0; r < 8; ++r) av[r] = alpha_s[16*wr + (lhi ? 8 : 0) + r];
#pragma unroll
        for (int t4 = 0; t4 < 4; ++t4)
#pragma unroll
            for (int r = 0; r < 8; ++r) o[t4][r] *= av[r];
#pragma unroll
        for (int kc = 0; kc < 64; kc += 32) {  // O += P @ V
            Frag a;
            const unsigned short* pa = Pb + (16*wr + l15) * 72 + kc + lhi * 8;
            a.q[0] = *(const uint4*)pa;  a.q[1] = *(const uint4*)(pa + 16);
#pragma unroll
            for (int t4 = 0; t4 < 4; ++t4) {
                Frag b;
                const unsigned short* pb = Vt + (64*wc + 16*t4 + l15) * 72 + kc + lhi * 8;
                b.q[0] = *(const uint4*)pb;  b.q[1] = *(const uint4*)(pb + 16);
                o[t4] = __builtin_amdgcn_wmma_f32_16x16x32_bf16(false, a.v, false, b.v, (short)0, o[t4], false, false);
            }
        }
    }
    __syncthreads();
    if (tid < 64) linv_s[tid] = 1.0f / l_i;
    __syncthreads();
    float lv[8];
#pragma unroll
    for (int r = 0; r < 8; ++r) lv[r] = linv_s[16*wr + (lhi ? 8 : 0) + r];
#pragma unroll
    for (int t4 = 0; t4 < 4; ++t4) {
        int n = 64*wc + 16*t4 + l15;
#pragma unroll
        for (int r = 0; r < 8; ++r) {
            int m = 16*wr + (lhi ? 8 : 0) + r;
            O[((size_t)(i0 + m) * 16 + h) * 128 + n] = f2bf(o[t4][r] * lv[r]);
        }
    }
}

// ---------------------------------------------------------------------------
// h = silu(g) * u (elementwise, vectorized)
// ---------------------------------------------------------------------------
__global__ __launch_bounds__(256)
void silu_mul_kernel(const unsigned short* __restrict__ g,
                     const unsigned short* __restrict__ u,
                     unsigned short* __restrict__ out) {
    size_t i = ((size_t)blockIdx.x * 256 + threadIdx.x) * 8;
    float gf[8], uf[8];
    unpack8(*(const uint4*)(g + i), gf);
    unpack8(*(const uint4*)(u + i), uf);
    unsigned short o[8];
#pragma unroll
    for (int r = 0; r < 8; ++r) {
        float s = gf[r] / (1.0f + __expf(-gf[r]));
        o[r] = f2bf(s * uf[r]);
    }
    *(uint4*)(out + i) = pack8(o);
}

// ---------------------------------------------------------------------------
extern "C" void kernel_launch(void* const* d_in, const int* in_sizes, int n_in,
                              void* d_out, int out_size, void* d_ws, size_t ws_size,
                              hipStream_t stream) {
    const int t = 4096, H = 2048, d = 128, qh = 16, kh = 4, F = 8192;
    const unsigned short* x     = (const unsigned short*)d_in[0];
    const float* sinp           = (const float*)d_in[1];
    const float* cosp           = (const float*)d_in[2];
    const float* pre_gamma      = (const float*)d_in[5];
    const float* q_gamma        = (const float*)d_in[6];
    const float* k_gamma        = (const float*)d_in[7];
    const float* post_gamma     = (const float*)d_in[8];
    const unsigned short* wq    = (const unsigned short*)d_in[9];
    const unsigned short* wk    = (const unsigned short*)d_in[10];
    const unsigned short* wv    = (const unsigned short*)d_in[11];
    const unsigned short* wo    = (const unsigned short*)d_in[12];
    const unsigned short* wgate = (const unsigned short*)d_in[13];
    const unsigned short* wup   = (const unsigned short*)d_in[14];
    const unsigned short* wdown = (const unsigned short*)d_in[15];

    char* ws = (char*)d_ws;
    size_t off = 0;
    auto alloc = [&](size_t bytes) -> char* {
        char* p = ws + off;
        off += (bytes + 255) & ~(size_t)255;
        return p;
    };
    unsigned short* xnorm = (unsigned short*)alloc((size_t)t * H * 2);
    unsigned short* qbuf  = (unsigned short*)alloc((size_t)t * qh * d * 2);
    unsigned short* kraw  = (unsigned short*)alloc((size_t)t * kh * d * 2);
    unsigned short* kbf   = (unsigned short*)alloc((size_t)t * kh * d * 2);
    unsigned short* attn  = (unsigned short*)alloc((size_t)t * qh * d * 2);
    unsigned short* x1    = (unsigned short*)alloc((size_t)t * H * 2);
    unsigned short* xn2   = (unsigned short*)alloc((size_t)t * H * 2);
    unsigned short* gbuf  = (unsigned short*)alloc((size_t)t * F * 2);
    unsigned short* ubuf  = (unsigned short*)alloc((size_t)t * F * 2);
    unsigned short* hbuf  = (unsigned short*)alloc((size_t)t * F * 2);

    // d_out = [x bf16 | k f32 | v bf16] in return order
    unsigned short* out_x = (unsigned short*)d_out;
    float*          out_k = (float*)((char*)d_out + (size_t)t * H * 2);
    unsigned short* out_v = (unsigned short*)((char*)d_out + (size_t)t * H * 2
                                              + (size_t)t * kh * d * 4);

    rmsnorm2048_kernel<<<t, 256, 0, stream>>>(x, pre_gamma, xnorm);

    gemm_bf16<false><<<dim3(qh * d / 128, t / 256), 256, 0, stream>>>(
        xnorm, wq, nullptr, qbuf, t, qh * d, H);
    gemm_bf16<false><<<dim3(kh * d / 128, t / 256), 256, 0, stream>>>(
        xnorm, wk, nullptr, kraw, t, kh * d, H);
    gemm_bf16<false><<<dim3(kh * d / 128, t / 256), 256, 0, stream>>>(
        xnorm, wv, nullptr, out_v, t, kh * d, H);

    rope_rmsnorm_kernel<<<t * qh, 128, 0, stream>>>(
        qbuf, q_gamma, sinp, cosp, qbuf, nullptr, qh);
    rope_rmsnorm_kernel<<<t * kh, 128, 0, stream>>>(
        kraw, k_gamma, sinp, cosp, kbf, out_k, kh);

    flash_attn_kernel<<<dim3(t / 64, qh), 256, 0, stream>>>(qbuf, kbf, out_v, attn);

    gemm_bf16<true><<<dim3(H / 128, t / 256), 256, 0, stream>>>(
        attn, wo, x, x1, t, H, qh * d);

    rmsnorm2048_kernel<<<t, 256, 0, stream>>>(x1, post_gamma, xn2);

    gemm_bf16<false><<<dim3(F / 128, t / 256), 256, 0, stream>>>(
        xn2, wgate, nullptr, gbuf, t, F, H);
    gemm_bf16<false><<<dim3(F / 128, t / 256), 256, 0, stream>>>(
        xn2, wup, nullptr, ubuf, t, F, H);

    silu_mul_kernel<<<(t * F) / (256 * 8), 256, 0, stream>>>(gbuf, ubuf, hbuf);

    gemm_bf16<true><<<dim3(H / 128, t / 256), 256, 0, stream>>>(
        hbuf, wdown, x1, out_x, t, H, F);
}
